// WindowAttention_PGF_46256797778038
// MI455X (gfx1250) — matrix operand order repaired and checked by reading, not compiled, not run on hardware
//
#include <hip/hip_runtime.h>

typedef __attribute__((ext_vector_type(16))) _Float16 v16h;
typedef __attribute__((ext_vector_type(8)))  float    v8f;
typedef __attribute__((ext_vector_type(4)))  unsigned int v4u;
typedef __attribute__((ext_vector_type(8)))  int      v8i;
typedef __attribute__((ext_vector_type(4)))  int      v4i;

// ---------------------------------------------------------------------------
// WMMA helpers (CDNA5 gfx1250, wave32). Layouts per CDNA5 ISA 7.12.2:
//  A (f16, 16x32): lane r=l&15 is row M; half=l>>4 selects K groups
//                  {half*8..+7} and {16+half*8..+7}.
//  B (f16, 32x16): lane n=l&15 is column N; half selects K groups {half*8..+7}, {16+half*8..+7}
//                  (same k order as A; source is [N][K] "transposed").
//  C/D (f32 16x16): c[j] is (M = j + 8*half, N = l&15).
// ---------------------------------------------------------------------------
__device__ __forceinline__ v8f wmma16x16x32(v16h a, v16h b, v8f c) {
  return __builtin_amdgcn_wmma_f32_16x16x32_f16(false, a, false, b,
                                                (short)0, c, false, false);
}

__device__ __forceinline__ v16h load_a16(const _Float16* tile, int ld) {
  const int l  = threadIdx.x & 31;
  const int rr = l & 15;
  const int hf = l >> 4;
  const _Float16* p = tile + rr * ld + hf * 8;
  v16h a;
#pragma unroll
  for (int i = 0; i < 8; ++i) { a[i] = p[i]; a[8 + i] = p[16 + i]; }
  return a;
}

__device__ __forceinline__ v16h load_bt16(const _Float16* tileT, int ld) {
  const int l  = threadIdx.x & 31;
  const int nn = l & 15;
  const int hf = l >> 4;
  const _Float16* p = tileT + nn * ld + hf * 8;
  v16h b;
#pragma unroll
  for (int i = 0; i < 8; ++i) { b[i] = p[i]; b[8 + i] = p[16 + i]; }
  return b;
}

// ---------------------------------------------------------------------------
// TDM: DMA a 2D f16 tile [rows x 32], row stride 32 (contiguous), from
// global memory into LDS.  D# packed per CDNA5 ISA ch.8:
//  group0: count=1 | lds_addr | global_addr[56:0] | type=2
//  group1: data_size=1 (2B), tensor_dim0=32, tensor_dim1=rows,
//          tile_dim0=32, tile_dim1=rows, tensor_dim0_stride=32
//  groups 2/3 (+ extended group): zero (2D tensor)
// This toolchain exposes the 6-arg builtin:
//  (uint32x4, int32x8, int32x4, int32x4, int32x8, i32 cpol)
// ---------------------------------------------------------------------------
__device__ __forceinline__ void tdm_load_rows32_f16(const _Float16* gsrc,
                                                    unsigned lds_byte_addr,
                                                    int rows) {
  unsigned long long ga = (unsigned long long)(__SIZE_TYPE__)gsrc;
  v4u g0;
  g0[0] = 1u;                                           // count=1 (valid user D#)
  g0[1] = lds_byte_addr;                                // lds_addr
  g0[2] = (unsigned)(ga & 0xFFFFFFFFu);                 // global_addr[31:0]
  g0[3] = (unsigned)((ga >> 32) & 0x01FFFFFFu)          // global_addr[56:32]
          | (2u << 30);                                 // type=2 ("image")
  v8i g1;
  g1[0] = (int)(1u << 16);               // wg_mask=0, data_size=1 (2 bytes)
  g1[1] = (int)(32u << 16);              // tensor_dim0[15:0]=32  (bits 63:48)
  g1[2] = (int)((unsigned)rows << 16);   // tensor_dim0 hi=0, tensor_dim1 lo=rows
  g1[3] = (int)(32u << 16);              // tensor_dim1 hi=0, tile_dim0=32
  g1[4] = rows;                          // tile_dim1=rows, tile_dim2=0
  g1[5] = 32;                            // tensor_dim0_stride[31:0]=32
  g1[6] = 0;                             // stride0 hi, stride1 lo
  g1[7] = 0;                             // stride1 hi
  v4i gz4 = {0, 0, 0, 0};
  v8i gz8 = {0, 0, 0, 0, 0, 0, 0, 0};
  __builtin_amdgcn_tensor_load_to_lds(g0, g1, gz4, gz4, gz8, 0);
}

// ---------------------------------------------------------------------------
// Kernel 1: kv projection GEMM.  skip[87808x384] @ kv_w[384x768] + kv_b.
// Epilogue splits k / v into [b,h,n,d] f16 buffers (k normalized later).
// Block = 256 thr (8 waves), tile 128(M) x 64(N), K step 32.
// ---------------------------------------------------------------------------
__global__ void gemm_kv_kernel(const float* __restrict__ skip,
                               const float* __restrict__ kv_w,
                               const float* __restrict__ kv_b,
                               _Float16* __restrict__ kbuf,
                               _Float16* __restrict__ vbuf) {
  __shared__ _Float16 aS[128 * 32];  // [m][k]
  __shared__ _Float16 bT[64 * 32];   // [n][k]
  const int t    = threadIdx.x;
  const int lane = t & 31;
  const int wave = t >> 5;
  const int wm = wave & 3, wn = wave >> 2;
  const int mbase = blockIdx.x * 128;
  const int nbase = blockIdx.y * 64;

  const v8f zero = {0.f, 0.f, 0.f, 0.f, 0.f, 0.f, 0.f, 0.f};
  v8f c[2][2];
  c[0][0] = zero; c[0][1] = zero; c[1][0] = zero; c[1][1] = zero;

  for (int kk = 0; kk < 384; kk += 32) {
    if (kk + 32 < 384)
      __builtin_prefetch(&skip[(size_t)(mbase + (t >> 1)) * 384 + kk + 32], 0, 1);
#pragma unroll
    for (int i = 0; i < 16; ++i) {
      int idx = i * 256 + t;           // 0..4095
      int r = idx >> 5, cc = idx & 31;
      aS[idx] = (_Float16)skip[(size_t)(mbase + r) * 384 + kk + cc];
    }
#pragma unroll
    for (int i = 0; i < 8; ++i) {
      int idx = i * 256 + t;           // 0..2047
      int kr = idx >> 6, n = idx & 63;
      bT[n * 32 + kr] = (_Float16)kv_w[(size_t)(kk + kr) * 768 + nbase + n];
    }
    __syncthreads();
    v16h a0 = load_a16(&aS[(wm * 32) * 32], 32);
    v16h a1 = load_a16(&aS[(wm * 32 + 16) * 32], 32);
    v16h b0 = load_bt16(&bT[(wn * 32) * 32], 32);
    v16h b1 = load_bt16(&bT[(wn * 32 + 16) * 32], 32);
    c[0][0] = wmma16x16x32(a0, b0, c[0][0]);
    c[0][1] = wmma16x16x32(a0, b1, c[0][1]);
    c[1][0] = wmma16x16x32(a1, b0, c[1][0]);
    c[1][1] = wmma16x16x32(a1, b1, c[1][1]);
    __syncthreads();
  }

  const int hf = lane >> 4;
  const int nn = lane & 15;
#pragma unroll
  for (int mt = 0; mt < 2; ++mt)
#pragma unroll
    for (int nt = 0; nt < 2; ++nt) {
      int ngl = nbase + wn * 32 + nt * 16 + nn;
      float bias = kv_b[ngl];
#pragma unroll
      for (int j = 0; j < 8; ++j) {
        int m = mbase + wm * 32 + mt * 16 + j + hf * 8;
        float val = c[mt][nt][j] + bias;
        int bb = m / 343, ntok = m - bb * 343;
        if (ngl < 384) {
          int h = ngl >> 5, dd = ngl & 31;
          kbuf[(((size_t)bb * 12 + h) * 343 + ntok) * 32 + dd] = (_Float16)val;
        } else {
          int cc2 = ngl - 384;
          int h = cc2 >> 5, dd = cc2 & 31;
          vbuf[(((size_t)bb * 12 + h) * 343 + ntok) * 32 + dd] = (_Float16)val;
        }
      }
    }
}

// ---------------------------------------------------------------------------
// Kernel 2: per-row (length d=32 == one wave32) cosine normalization.
// Builds qn from x_up and normalizes k in place.  One wave per (b,h,n) row.
// ---------------------------------------------------------------------------
__global__ void prep_kernel(const float* __restrict__ x_up,
                            _Float16* __restrict__ qn,
                            _Float16* __restrict__ kbuf) {
  const size_t row = (size_t)blockIdx.x * 8 + (threadIdx.x >> 5);
  const int dd = threadIdx.x & 31;
  const size_t NROWS = (size_t)3072 * 343;
  if (row >= NROWS) return;
  int b   = (int)(row / (12 * 343));
  int rem = (int)(row - (size_t)b * (12 * 343));
  int h = rem / 343;
  int n = rem - h * 343;

  // q normalization (from x_up, layout [b,n,h*32+dd])
  float x = x_up[((size_t)b * 343 + n) * 384 + h * 32 + dd];
  float ss = x * x;
#pragma unroll
  for (int off = 16; off; off >>= 1) ss += __shfl_xor(ss, off, 32);
  float inv = 1.0f / fmaxf(sqrtf(ss), 1e-12f);
  qn[row * 32 + dd] = (_Float16)(x * inv);

  // k normalization in place
  float kx = (float)kbuf[row * 32 + dd];
  float ks = kx * kx;
#pragma unroll
  for (int off = 16; off; off >>= 1) ks += __shfl_xor(ks, off, 32);
  float ki = 1.0f / fmaxf(sqrtf(ks), 1e-12f);
  kbuf[row * 32 + dd] = (_Float16)(kx * ki);
}

// ---------------------------------------------------------------------------
// Kernel 3: fused window attention per (b,h).  4 waves x 16 queries = 64
// queries per block.  K [352x32] staged via the Tensor Data Mover (TDM),
// V transposed into [32x352] by the threads, rel-pos-bias computed
// analytically (7x7x7 window), softmax via half-wave shuffles, P staged
// per-wave in LDS for the PV A-fragments.
// ---------------------------------------------------------------------------
__global__ void attn_kernel(const _Float16* __restrict__ qn,
                            const _Float16* __restrict__ kbuf,
                            const _Float16* __restrict__ vbuf,
                            const float* __restrict__ mask,
                            const float* __restrict__ logit_scale,
                            const float* __restrict__ rpb,
                            _Float16* __restrict__ attnout) {
  extern __shared__ __align__(16) char smem[];
  _Float16* kS = (_Float16*)smem;          // [352][32]  key-major
  _Float16* vT = kS + 352 * 32;            // [32][352]  d-major (transposed)
  _Float16* pS = vT + 32 * 352;            // 4 x [16][352] per-wave P staging

  const int t    = threadIdx.x;            // 0..127
  const int lane = t & 31;
  const int wave = t >> 5;
  const int bh = blockIdx.x;
  const int b = bh / 12, h = bh - b * 12;
  const int qbase = blockIdx.y * 64 + wave * 16;
  const size_t kvbase = (size_t)bh * 343 * 32;

  // TDM: async DMA the contiguous K tile [343 x 32] f16 into kS (wave 0 only;
  // TDM ignores EXEC and issues once per wave).
  if (wave == 0)
    tdm_load_rows32_f16(kbuf + kvbase, (unsigned)(__SIZE_TYPE__)kS, 343);

  // Threads stage V transposed and zero the 9 padded key rows of kS.
#pragma unroll 4
  for (int i = 0; i < 88; ++i) {
    int idx = i * 128 + t;                 // 0..11263
    int key = idx >> 5, dd = idx & 31;
    _Float16 vv = (key < 343) ? vbuf[kvbase + (size_t)key * 32 + dd]
                              : (_Float16)0.0f;
    vT[dd * 352 + key] = vv;
  }
#pragma unroll
  for (int i = 0; i < 3; ++i) {
    int idx = 343 * 32 + i * 128 + t;      // pad region: 288 halves
    if (idx < 352 * 32) kS[idx] = (_Float16)0.0f;
  }
  if (wave == 0) __builtin_amdgcn_s_wait_tensorcnt(0);
  __syncthreads();

  const float scale = __expf(fminf(logit_scale[h], 4.6051701859880913f));

  // Q fragment (A layout), query rows clamped for padded tiles
  const int rr = lane & 15;
  const int hf = lane >> 4;
  int qrow = qbase + rr; if (qrow > 342) qrow = 342;
  const _Float16* qp = qn + kvbase + (size_t)qrow * 32;
  v16h qa;
#pragma unroll
  for (int i = 0; i < 8; ++i) { qa[i] = qp[hf * 8 + i]; qa[8 + i] = qp[16 + hf * 8 + i]; }

  // S = Qn x Kn^T : 22 key tiles, K-dim = d = 32 (one WMMA each)
  const v8f zero = {0.f, 0.f, 0.f, 0.f, 0.f, 0.f, 0.f, 0.f};
  v8f s[22];
#pragma unroll
  for (int kt = 0; kt < 22; ++kt) {
    v16h bf = load_bt16(&kS[(kt * 16) * 32], 32);
    s[kt] = wmma16x16x32(qa, bf, zero);
  }

  // epilogue: scale + rel-pos-bias + shift-window mask (D layout indexing)
  const int w = b & 63;  // b % nW
#pragma unroll
  for (int kt = 0; kt < 22; ++kt) {
    int key = kt * 16 + rr;
    int kz = key / 49, kr2 = key - kz * 49;
    int ky = kr2 / 7,  kx  = kr2 - ky * 7;
#pragma unroll
    for (int j = 0; j < 8; ++j) {
      int qi = qbase + j + hf * 8;
      int qc = qi > 342 ? 342 : qi;
      float sv;
      if (key < 343) {
        int qz = qc / 49, qr2 = qc - qz * 49;
        int qy = qr2 / 7, qx  = qr2 - qy * 7;
        int idx = (qz - kz + 6) * 20 + (qy - ky + 6) * 13 + (qx - kx + 6);
        float bias = rpb[idx * 12 + h];
        float mk = mask[((size_t)w * 343 + qc) * 343 + key];
        sv = s[kt][j] * scale + bias + mk;
      } else {
        sv = -3.0e38f;                 // padded key -> prob 0
      }
      s[kt][j] = sv;
    }
  }

  // softmax over 343 keys: per-row max/sum, rows live in 16-lane halves
#pragma unroll
  for (int j = 0; j < 8; ++j) {
    float m = -3.4e38f;
#pragma unroll
    for (int kt = 0; kt < 22; ++kt) m = fmaxf(m, s[kt][j]);
#pragma unroll
    for (int off = 1; off < 16; off <<= 1) m = fmaxf(m, __shfl_xor(m, off, 32));
    float sum = 0.f;
#pragma unroll
    for (int kt = 0; kt < 22; ++kt) { float p = __expf(s[kt][j] - m); s[kt][j] = p; sum += p; }
#pragma unroll
    for (int off = 1; off < 16; off <<= 1) sum += __shfl_xor(sum, off, 32);
    float is = 1.0f / sum;
#pragma unroll
    for (int kt = 0; kt < 22; ++kt) s[kt][j] *= is;
  }

  // stage P (f16) per-wave, rearranging D layout -> row-major [16][352]
  _Float16* pw = pS + wave * 16 * 352;
#pragma unroll
  for (int kt = 0; kt < 22; ++kt)
#pragma unroll
    for (int j = 0; j < 8; ++j)
      pw[(j + hf * 8) * 352 + kt * 16 + rr] = (_Float16)s[kt][j];
  __syncthreads();

  // O = P x V : 11 K-chunks of 32 keys, 2 output d-tiles
  v8f o0 = zero, o1 = zero;
#pragma unroll
  for (int kc = 0; kc < 11; ++kc) {
    v16h pa = load_a16(pw + kc * 32, 352);
    v16h b0 = load_bt16(&vT[0 * 352 + kc * 32], 352);
    v16h b1 = load_bt16(&vT[16 * 352 + kc * 32], 352);
    o0 = wmma16x16x32(pa, b0, o0);
    o1 = wmma16x16x32(pa, b1, o1);
  }

#pragma unroll
  for (int j = 0; j < 8; ++j) {
    int qi = qbase + j + hf * 8;
    if (qi < 343) {
      size_t base = ((size_t)b * 343 + qi) * 384 + h * 32;
      attnout[base + rr]      = (_Float16)o0[j];
      attnout[base + 16 + rr] = (_Float16)o1[j];
    }
  }
}

// ---------------------------------------------------------------------------
// Kernel 4: output projection GEMM. (attnout + pos_embed)[87808x384] @
// proj_w[384x384] + proj_b -> d_out (f32).  pos_embed fused into A staging.
// ---------------------------------------------------------------------------
__global__ void gemm_proj_kernel(const _Float16* __restrict__ attn,
                                 const float* __restrict__ pos,
                                 const float* __restrict__ proj_w,
                                 const float* __restrict__ proj_b,
                                 float* __restrict__ out) {
  __shared__ _Float16 aS[128 * 32];
  __shared__ _Float16 bT[64 * 32];
  const int t    = threadIdx.x;
  const int lane = t & 31;
  const int wave = t >> 5;
  const int wm = wave & 3, wn = wave >> 2;
  const int mbase = blockIdx.x * 128;
  const int nbase = blockIdx.y * 64;

  const v8f zero = {0.f, 0.f, 0.f, 0.f, 0.f, 0.f, 0.f, 0.f};
  v8f c[2][2];
  c[0][0] = zero; c[0][1] = zero; c[1][0] = zero; c[1][1] = zero;

  for (int kk = 0; kk < 384; kk += 32) {
#pragma unroll
    for (int i = 0; i < 16; ++i) {
      int idx = i * 256 + t;
      int r = idx >> 5, cc = idx & 31;
      size_t g = (size_t)(mbase + r) * 384 + kk + cc;
      aS[idx] = (_Float16)((float)attn[g] + pos[g]);
    }
#pragma unroll
    for (int i = 0; i < 8; ++i) {
      int idx = i * 256 + t;
      int kr = idx >> 6, n = idx & 63;
      bT[n * 32 + kr] = (_Float16)proj_w[(size_t)(kk + kr) * 384 + nbase + n];
    }
    __syncthreads();
    v16h a0 = load_a16(&aS[(wm * 32) * 32], 32);
    v16h a1 = load_a16(&aS[(wm * 32 + 16) * 32], 32);
    v16h b0 = load_bt16(&bT[(wn * 32) * 32], 32);
    v16h b1 = load_bt16(&bT[(wn * 32 + 16) * 32], 32);
    c[0][0] = wmma16x16x32(a0, b0, c[0][0]);
    c[0][1] = wmma16x16x32(a0, b1, c[0][1]);
    c[1][0] = wmma16x16x32(a1, b0, c[1][0]);
    c[1][1] = wmma16x16x32(a1, b1, c[1][1]);
    __syncthreads();
  }

  const int hf = lane >> 4;
  const int nn = lane & 15;
#pragma unroll
  for (int mt = 0; mt < 2; ++mt)
#pragma unroll
    for (int nt = 0; nt < 2; ++nt) {
      int ngl = nbase + wn * 32 + nt * 16 + nn;
      float bias = proj_b[ngl];
#pragma unroll
      for (int j = 0; j < 8; ++j) {
        int m = mbase + wm * 32 + mt * 16 + j + hf * 8;
        out[(size_t)m * 384 + ngl] = c[mt][nt][j] + bias;
      }
    }
}

// ---------------------------------------------------------------------------
extern "C" void kernel_launch(void* const* d_in, const int* in_sizes, int n_in,
                              void* d_out, int out_size, void* d_ws, size_t ws_size,
                              hipStream_t stream) {
  const float* skip        = (const float*)d_in[0];
  const float* x_up        = (const float*)d_in[1];
  const float* pos_embed   = (const float*)d_in[2];
  const float* mask        = (const float*)d_in[3];
  const float* kv_w        = (const float*)d_in[4];
  const float* kv_b        = (const float*)d_in[5];
  const float* proj_w      = (const float*)d_in[6];
  const float* proj_b      = (const float*)d_in[7];
  const float* logit_scale = (const float*)d_in[8];
  const float* rpb         = (const float*)d_in[9];
  float* out = (float*)d_out;

  // workspace: qn | kn | v | attn-out, each 3072*343*32 f16 = 67,436,544 B
  char* ws = (char*)d_ws;
  const size_t SZ = 67436544ull;
  _Float16* qn = (_Float16*)(ws + 0 * SZ);
  _Float16* kn = (_Float16*)(ws + 1 * SZ);
  _Float16* vv = (_Float16*)(ws + 2 * SZ);
  _Float16* ao = (_Float16*)(ws + 3 * SZ);

  // 1) kv projection GEMM  (M=87808=128*686, N=768, K=384)
  gemm_kv_kernel<<<dim3(686, 12), 256, 0, stream>>>(skip, kv_w, kv_b, kn, vv);
  // 2) cosine normalization of q (from x_up) and k (in place)
  prep_kernel<<<131712, 256, 0, stream>>>(x_up, qn, kn);
  // 3) fused attention: grid (b*h, query-tiles of 64); 88.0 KB dynamic LDS
  attn_kernel<<<dim3(3072, 6), 128, 90112, stream>>>(qn, kn, vv, mask,
                                                     logit_scale, rpb, ao);
  // 4) output projection GEMM (M=87808, N=384, K=384), pos_embed fused
  gemm_proj_kernel<<<dim3(686, 6), 256, 0, stream>>>(ao, pos_embed,
                                                     proj_w, proj_b, out);
}
